// Codebook_77962246357406
// MI455X (gfx1250) — matrix operand order repaired and checked by reading, not compiled, not run on hardware
//
#include <hip/hip_runtime.h>
#include <hip/hip_bf16.h>

// ---------------------------------------------------------------------------
// VQ-VAE Codebook eval forward, fused for MI455X (gfx1250, wave32, WMMA).
//   distances argmin via  ||e||^2 - 2 x.e   (||x||^2 dropped: constant per row)
//   GEMM in bf16 WMMA (v_wmma_f32_16x16x32_bf16), everything else fp32.
// ---------------------------------------------------------------------------

typedef __attribute__((ext_vector_type(16))) __bf16 v16bf;
typedef __attribute__((ext_vector_type(8)))  float  v8f;

#define N_CODES   2048
#define DIMK      256
#define THW       16384          // 16*32*32 tokens per batch element
#define N_TOKENS  65536          // 4*THW
#define N_ELEMS   16777216       // 4*256*THW

static __device__ __forceinline__ unsigned short f2bf(float f) {
    union { float f; unsigned u; } x; x.f = f;
    unsigned u = x.u;
    u += 0x7fffu + ((u >> 16) & 1u);          // round-to-nearest-even
    return (unsigned short)(u >> 16);
}

// ---------------------------------------------------------------------------
// Kernel 1: embeddings fp32 -> bf16 row-major table + ||e||^2 per code
// ---------------------------------------------------------------------------
__global__ void __launch_bounds__(256)
vq_prep_kernel(const float* __restrict__ emb,
               unsigned short* __restrict__ emb_bf,
               float* __restrict__ enorm) {
    const int code = blockIdx.x;
    const int c    = threadIdx.x;
    const float v  = emb[code * DIMK + c];
    emb_bf[code * DIMK + c] = f2bf(v);

    __shared__ float red[256];
    red[c] = v * v;
    __syncthreads();
    for (int st = 128; st > 0; st >>= 1) {
        if (c < st) red[c] += red[c + st];
        __syncthreads();
    }
    if (c == 0) enorm[code] = red[0];
}

// ---------------------------------------------------------------------------
// Kernel 2: per-token argmin over 2048 codes via bf16 WMMA GEMM.
//   grid = 512 blocks x 256 threads (8 waves); wave handles 16 tokens.
//   A = 16x32 code tile (from LDS, double buffered), B = 32x16 token tile
//   (registers, all K=256 resident), D = 16(code) x 16(token) f32.
// ---------------------------------------------------------------------------
__global__ void __launch_bounds__(256)
vq_argmin_kernel(const float* __restrict__ z,
                 const unsigned short* __restrict__ emb_bf,
                 const float* __restrict__ enorm,
                 int* __restrict__ idxws,
                 float* __restrict__ out_idx,
                 unsigned* __restrict__ counts) {
    __shared__ __align__(16) unsigned char smem[2][8192];   // 16 codes x 256 K bf16, x2

    const int tid    = threadIdx.x;
    const int lane   = tid & 31;
    const int wave   = tid >> 5;
    const int lane16 = lane & 15;
    const int hi     = lane >> 4;                 // lane half: 0 or 1
    const int tb     = blockIdx.x * 128 + wave * 16;

    // ---- load this wave's 16 tokens (B matrix) as bf16, all K=256 ----------
    // z layout [B, C, T, H, W]: elem (token, c) at b*256*THW + c*THW + s
    const int token = tb + lane16;
    const int b     = token >> 14;
    const int s     = token & (THW - 1);
    const float* zbase = z + (size_t)b * DIMK * THW + s;

    v16bf bmat[8];
    #pragma unroll
    for (int kc = 0; kc < 8; ++kc) {
        union { v16bf v; unsigned short h[16]; } u;
        const int koff = kc * 32 + hi * 16;       // B layout: lanes 0-15 K=0..15, 16-31 K=16..31
        #pragma unroll
        for (int e = 0; e < 16; ++e)
            u.h[e] = f2bf(zbase[(size_t)(koff + e) * THW]);
        bmat[kc] = u.v;
    }

    // ---- stage code tile 0 into LDS ---------------------------------------
    const uint4* gtab = (const uint4*)emb_bf;     // 128 tiles x 512 uint4
    uint4 pf0 = gtab[tid * 2];
    uint4 pf1 = gtab[tid * 2 + 1];
    ((uint4*)smem[0])[tid * 2]     = pf0;
    ((uint4*)smem[0])[tid * 2 + 1] = pf1;

    float minv = 3.402823466e38f;
    int   mini = 0;

    for (int ct = 0; ct < 128; ++ct) {
        const int cur = ct & 1;
        if (ct + 1 < 128) {                       // prefetch next tile to regs
            pf0 = gtab[(ct + 1) * 512 + tid * 2];
            pf1 = gtab[(ct + 1) * 512 + tid * 2 + 1];
            if (ct + 2 < 128)
                __builtin_prefetch(gtab + (ct + 2) * 512 + tid * 2, 0, 1);
        }
        __syncthreads();                          // tile `cur` ready in LDS

        // A-tile row for this lane: M = lane16, K-half by lane half (bytes)
        const unsigned char* sb = smem[cur];
        const int arow = lane16 * 512 + hi * 32;

        v8f acc = {};
        #pragma unroll
        for (int kc = 0; kc < 8; ++kc) {
            union { v16bf v; uint4 q[2]; } ua;
            const uint4* ap = (const uint4*)(sb + arow + kc * 64);
            ua.q[0] = ap[0];                      // K = kb+0..7
            ua.q[1] = ap[1];                      // K = kb+8..15 (+16 bytes)
            acc = __builtin_amdgcn_wmma_f32_16x16x32_bf16(
                false, ua.v, false, bmat[kc], (short)0, acc, false, false);
        }

        // score = ||e||^2 - 2 x.e ; lane's 8 codes = 16*ct + 8*hi + r
        const float* ep = enorm + ct * 16 + hi * 8;
        #pragma unroll
        for (int r = 0; r < 8; ++r) {
            const float sc   = fmaf(-2.0f, acc[r], ep[r]);
            const int   code = ct * 16 + hi * 8 + r;
            if (sc < minv) { minv = sc; mini = code; }
        }

        __syncthreads();                          // all reads of `cur` done
        if (ct + 1 < 128) {
            uint4* d = (uint4*)smem[(ct + 1) & 1];
            d[tid * 2]     = pf0;
            d[tid * 2 + 1] = pf1;
        }
    }

    // ---- combine lane halves (codes r vs r+8 live in lanes l and l+16) ----
    const float ov = __shfl_xor(minv, 16, 32);
    const int   oi = __shfl_xor(mini, 16, 32);
    if (ov < minv || (ov == minv && oi < mini)) { minv = ov; mini = oi; }

    if (lane < 16) {
        const int t = tb + lane;
        idxws[t]   = mini;
        out_idx[t] = (float)mini;
        atomicAdd(&counts[mini], 1u);
    }
}

// ---------------------------------------------------------------------------
// Kernel 3: gather fp32 embedding into [B,C,T,H,W] output + loss accumulation
// ---------------------------------------------------------------------------
__global__ void __launch_bounds__(256)
vq_gather_loss_kernel(const float* __restrict__ z,
                      const float* __restrict__ emb,
                      const int* __restrict__ idxws,
                      float* __restrict__ out,
                      float* __restrict__ lossacc) {
    const unsigned tid = blockIdx.x * 256u + threadIdx.x;   // linear [B,C,T,H,W]
    const unsigned s   = tid & (THW - 1);
    const unsigned c   = (tid >> 14) & 255u;
    const unsigned b   = tid >> 22;
    const unsigned n   = (b << 14) | s;                     // token index

    const int   code = idxws[n];
    const float e    = emb[(unsigned)code * DIMK + c];
    const float zv   = z[tid];
    out[tid] = e;                                           // ST output == emb_q

    const float d  = zv - e;
    __shared__ float red[256];
    red[threadIdx.x] = d * d;
    __syncthreads();
    for (int st = 128; st > 0; st >>= 1) {
        if ((int)threadIdx.x < st) red[threadIdx.x] += red[threadIdx.x + st];
        __syncthreads();
    }
    if (threadIdx.x == 0) atomicAdd(lossacc, red[0]);
}

// ---------------------------------------------------------------------------
// Kernel 4: scalars (commitment loss, perplexity, codes used, usage ratio)
// ---------------------------------------------------------------------------
__global__ void __launch_bounds__(256)
vq_finalize_kernel(const unsigned* __restrict__ counts,
                   const float* __restrict__ lossacc,
                   float* __restrict__ scal) {
    __shared__ float rent[256];
    __shared__ int   rused[256];
    float ent = 0.0f;
    int used = 0;
    for (int i = threadIdx.x; i < N_CODES; i += 256) {
        const float p = (float)counts[i] * (1.0f / (float)N_TOKENS);
        ent  += p * __logf(p + 1e-7f);
        used += (p > 0.0f) ? 1 : 0;
    }
    rent[threadIdx.x]  = ent;
    rused[threadIdx.x] = used;
    __syncthreads();
    for (int st = 128; st > 0; st >>= 1) {
        if ((int)threadIdx.x < st) {
            rent[threadIdx.x]  += rent[threadIdx.x + st];
            rused[threadIdx.x] += rused[threadIdx.x + st];
        }
        __syncthreads();
    }
    if (threadIdx.x == 0) {
        float loss = lossacc[0] * (1.0f / (float)N_ELEMS);
        loss = fminf(loss, 10.0f) * 0.25f;
        scal[0] = loss;
        scal[1] = __expf(-rent[0]);
        scal[2] = (float)rused[0];
        scal[3] = (float)rused[0] * (1.0f / (float)N_CODES);
    }
}

// ---------------------------------------------------------------------------
extern "C" void kernel_launch(void* const* d_in, const int* in_sizes, int n_in,
                              void* d_out, int out_size, void* d_ws, size_t ws_size,
                              hipStream_t stream) {
    const float* z   = (const float*)d_in[0];   // [4,256,16,32,32] fp32
    const float* emb = (const float*)d_in[1];   // [2048,256] fp32
    float* out = (float*)d_out;

    // workspace layout
    char* ws = (char*)d_ws;
    unsigned short* emb_bf = (unsigned short*)ws;               // 1,048,576 B
    float*    enorm   = (float*)   (ws + 1048576);              //     8,192 B
    unsigned* counts  = (unsigned*)(ws + 1056768);              //     8,192 B
    float*    lossacc = (float*)   (ws + 1064960);              //        16 B
    int*      idxws   = (int*)     (ws + 1064976);              //   262,144 B

    float* out_q    = out;                 // [4,256,16,32,32] -> 16,777,216
    float* out_idx  = out + 16777216;      // [4,16,32,32]     ->     65,536
    float* out_scal = out + 16842752;      // 4 scalars

    hipMemsetAsync(counts, 0, 8192 + 16, stream);   // counts + loss accumulator

    vq_prep_kernel<<<N_CODES, 256, 0, stream>>>(emb, emb_bf, enorm);
    vq_argmin_kernel<<<N_TOKENS / 128, 256, 0, stream>>>(z, emb_bf, enorm,
                                                         idxws, out_idx, counts);
    vq_gather_loss_kernel<<<N_ELEMS / 256, 256, 0, stream>>>(z, emb, idxws,
                                                             out_q, lossacc);
    vq_finalize_kernel<<<1, 256, 0, stream>>>(counts, lossacc, out_scal);
}
